// HomographyEstimator_24678882083257
// MI455X (gfx1250) — matrix operand order repaired and checked by reading, not compiled, not run on hardware
//
#include <hip/hip_runtime.h>
#include <cstdint>
#include <cstddef>

typedef __attribute__((ext_vector_type(16))) _Float16 v16h;
typedef __attribute__((ext_vector_type(8)))  _Float16 v8h;
typedef __attribute__((ext_vector_type(8)))  float    v8f;
typedef __attribute__((ext_vector_type(4)))  int      v4i;

// ---------------------------------------------------------------------------
// Utility kernels
// ---------------------------------------------------------------------------
__global__ void k_zero_h(_Float16* __restrict__ p, int n) {
  int i = blockIdx.x * 256 + threadIdx.x;
  if (i < n) p[i] = (_Float16)0.f;
}

// gray image -> padded NHWC [130*130][32], channel0 = mean, rest stay zero
__global__ void k_gray(const float* __restrict__ in, _Float16* __restrict__ g) {
  int i = blockIdx.x * 256 + threadIdx.x;
  if (i >= 16384) return;
  int y = i >> 7, x = i & 127;
  g[((size_t)((y + 1) * 130 + (x + 1))) * 32] =
      (_Float16)((in[i] + in[16384 + i] + in[32768 + i]) * (1.0f / 3.0f));
}

// Fold w1a [64,3,3,3] over input channels -> f16 [64][9*32], k = s*32 + c
__global__ void k_fold(const float* __restrict__ w, _Float16* __restrict__ dst) {
  int i = blockIdx.x * 256 + threadIdx.x;   // 64*288
  if (i >= 64 * 288) return;
  int o = i / 288, k = i - o * 288;
  int s = k >> 5, c = k & 31;
  float v = 0.f;
  if (c == 0) v = w[o * 27 + s] + w[o * 27 + 9 + s] + w[o * 27 + 18 + s];
  dst[i] = (_Float16)v;
}

// Repack conv weights f32 [O][Cin][3][3] -> f16 [O][9*Cinp], k = s*Cinp + c
__global__ void k_cvt(const float* __restrict__ src, _Float16* __restrict__ dst,
                      int O, int Cin, int Cinp) {
  int i = blockIdx.x * 256 + threadIdx.x;
  int Kp = 9 * Cinp;
  if (i >= O * Kp) return;
  int o = i / Kp, k = i - o * Kp;
  int s = k / Cinp, c = k - s * Cinp;
  _Float16 v = (_Float16)0.f;
  if (c < Cin) v = (_Float16)src[((size_t)o * Cin + c) * 9 + s];
  dst[i] = v;
}

// 2x2 maxpool on padded NHWC: [2Ho+2][2Wo+2][C] -> [Ho+2][Wo+2][C] interior
__global__ void k_maxpool(const _Float16* __restrict__ in, _Float16* __restrict__ out,
                          int C, int Ho, int Wo) {
  int i = blockIdx.x * 256 + threadIdx.x;
  int tot = Ho * Wo * C;
  if (i >= tot) return;
  int c = i % C; int pix = i / C;
  int yo = pix / Wo, xo = pix - yo * Wo;
  int Wpi = 2 * Wo + 2, Wpo = Wo + 2;
  const _Float16* p = in + ((size_t)((2 * yo + 1) * Wpi + (2 * xo + 1))) * C + c;
  float a = (float)p[0], b = (float)p[C];
  float d = (float)p[(size_t)Wpi * C], e = (float)p[(size_t)Wpi * C + C];
  out[((size_t)((yo + 1) * Wpo + (xo + 1))) * C + c] = (_Float16)fmaxf(fmaxf(a, b), fmaxf(d, e));
}

// L2-normalize 64ch over padded 18x18 NHWC input.
// pad==0: write dense [256][64].  pad==1: write padded [48*48][64] at +16.
__global__ void k_l2norm(const _Float16* __restrict__ f, _Float16* __restrict__ dst, int pad) {
  int p = threadIdx.x;                        // 256 pixels
  int y = p >> 4, x = p & 15;
  const _Float16* src = f + ((size_t)((y + 1) * 18 + (x + 1))) * 64;
  float s = 0.f;
  for (int c = 0; c < 64; ++c) { float v = (float)src[c]; s += v * v; }
  float inv = 1.f / fmaxf(sqrtf(s), 1e-12f);
  _Float16* d = pad ? dst + ((size_t)((y + 16) * 48 + (x + 16))) * 64
                    : dst + (size_t)p * 64;
  for (int c = 0; c < 64; ++c) d[c] = (_Float16)((float)src[c] * inv);
}

// ---------------------------------------------------------------------------
// Cost volume: block = offset o (1089), thread = pixel p (256).
// fa ([256][64], 32KB) staged in LDS via async-to-LDS.
// Output: padded NHWC [18*18][1120] (pre-zeroed: halo + channel pad).
// ---------------------------------------------------------------------------
__global__ __launch_bounds__(256) void k_costvol(const _Float16* __restrict__ fa,
                                                 const _Float16* __restrict__ fbp, // [48*48][64]
                                                 _Float16* __restrict__ cv) {      // [18*18][1120]
  __shared__ __align__(16) _Float16 sfa[256 * 64];
  int t = threadIdx.x;
#if __has_builtin(__builtin_amdgcn_global_load_async_to_lds_b128)
  {
    typedef v4i __attribute__((address_space(1)))* gv4;
    typedef v4i __attribute__((address_space(3)))* lv4;
    const char* g = (const char*)fa + t * 128;
    char*       l = (char*)sfa + t * 128;
#pragma unroll
    for (int i = 0; i < 8; ++i)
      __builtin_amdgcn_global_load_async_to_lds_b128(
          (gv4)(uintptr_t)(g + 16 * i), (lv4)(uintptr_t)(l + 16 * i), 0, 0);
#if __has_builtin(__builtin_amdgcn_s_wait_asynccnt)
    __builtin_amdgcn_s_wait_asynccnt(0);
#endif
  }
#else
  {
    const float4* g4 = (const float4*)fa;
    float4*       l4 = (float4*)sfa;
#pragma unroll
    for (int i = 0; i < 8; ++i) l4[t * 8 + i] = g4[t * 8 + i];
  }
#endif
  __syncthreads();

  int o = blockIdx.x;                        // 0..1088
  int dy = o / 33, dx = o - dy * 33;
  int y = t >> 4, x = t & 15;
  const _Float16* fb = fbp + ((size_t)((y + dy) * 48 + (x + dx))) * 64;
  const _Float16* fl = sfa + (size_t)t * 64;
  float s = 0.f;
#pragma unroll
  for (int c = 0; c < 64; c += 8) {
    v8h va = *(const v8h*)(fl + c);
    v8h vb = *(const v8h*)(fb + c);
#pragma unroll
    for (int j = 0; j < 8; ++j) s += (float)va[j] * (float)vb[j];
  }
  float r = s * (1.f / 64.f);
  cv[((size_t)((y + 1) * 18 + (x + 1))) * 1120 + o] = (_Float16)(r > 0.f ? r : 0.1f * r);
}

// ---------------------------------------------------------------------------
// Implicit-GEMM 3x3 SAME conv via V_WMMA_F32_16X16X32_F16, fully vectorized.
//  in : f16 padded NHWC [(H+2)*(W+2)][Cin]   (zero halo, Cin multiple of 32)
//  wt : f16 [Cout][9*Cin], k = s*Cin + c  (spatial-major K)
//  A = 16 weight rows (M = out-channels), B = 16 pixels (N), K = 9*Cin.
//  outh: f16 padded NHWC [(H+2)*(W+2)][Cout]  (interior written)
//  outf: f32 NCHW [Cout][H*W]  (flatten order for the FC)
// ---------------------------------------------------------------------------
__global__ __launch_bounds__(256)
void k_conv3x3_wmma(const _Float16* __restrict__ in, const _Float16* __restrict__ wt,
                    const float* __restrict__ bias, _Float16* __restrict__ outh,
                    float* __restrict__ outf, int H, int W, int Cin, int Cout, int relu) {
  const int HW      = H * W;
  const int Wp      = W + 2;
  const int Kpad    = 9 * Cin;
  const int tiles_n = HW >> 4;               // pixel tiles
  const int tiles_m = Cout >> 4;             // channel tiles
  const int total   = tiles_m * tiles_n;
  const int wid     = blockIdx.x * 8 + (threadIdx.x >> 5);
  if (wid >= total) return;                  // wave-uniform
  const int lane = threadIdx.x & 31;
  const int lo   = lane & 15;
  const int half = lane >> 4;
  const int mt = wid / tiles_n, nt = wid - mt * tiles_n;

  const int p = nt * 16 + lo;                // this lane's pixel (B column)
  const int y = p / W, x = p - y * W;
  const _Float16* wrow = wt + (size_t)(mt * 16 + lo) * Kpad + 16 * half;  // A row

  v8f acc = {};
#pragma unroll
  for (int s = 0; s < 9; ++s) {
    const int ky = s / 3, kx = s - ky * 3;   // compile-time after unroll
    const _Float16* bbase = in + ((size_t)((y + ky) * Wp + (x + kx))) * Cin + 16 * half;
    const _Float16* abase = wrow + s * Cin;
    for (int c0 = 0; c0 < Cin; c0 += 32) {
      v8h a0 = *(const v8h*)(abase + c0);        // A: k = s*Cin+c0+16*half+0..7
      v8h a1 = *(const v8h*)(abase + c0 + 8);    //    k = ... +8+0..7
      v16h a = __builtin_shufflevector(a0, a1, 0, 1, 2, 3, 4, 5, 6, 7,
                                               8, 9, 10, 11, 12, 13, 14, 15);
      v16h b = *(const v16h*)(bbase + c0);       // B: 16 contiguous channels
      acc = __builtin_amdgcn_wmma_f32_16x16x32_f16(false, a, false, b, (short)0, acc,
                                                   false, false);
    }
  }

  const int   cbase = mt * 16 + 8 * half;    // D: VGPR r -> channel cbase + r
  const float4 bv0 = *(const float4*)(bias + cbase);
  const float4 bv1 = *(const float4*)(bias + cbase + 4);
  float v[8] = { acc[0] + bv0.x, acc[1] + bv0.y, acc[2] + bv0.z, acc[3] + bv0.w,
                 acc[4] + bv1.x, acc[5] + bv1.y, acc[6] + bv1.z, acc[7] + bv1.w };
  if (relu) {
#pragma unroll
    for (int r = 0; r < 8; ++r) v[r] = fmaxf(v[r], 0.f);
  }
  if (outh) {
    union { _Float16 h[8]; uint4 u; } pk;
#pragma unroll
    for (int r = 0; r < 8; ++r) pk.h[r] = (_Float16)v[r];
    *(uint4*)(outh + ((size_t)((y + 1) * Wp + (x + 1))) * Cout + cbase) = pk.u;
  } else {
#pragma unroll
    for (int r = 0; r < 8; ++r) outf[(size_t)(cbase + r) * HW + p] = v[r];
  }
}

// ---------------------------------------------------------------------------
// FC1: streams the 536MB f32 weight matrix -> the global bandwidth bound.
// Grid (16 k-chunks, 1024 rows); deterministic two-stage reduction.
// ---------------------------------------------------------------------------
__global__ __launch_bounds__(256) void k_fc1(const float* __restrict__ Wm,
                                             const float* __restrict__ xv,
                                             float* __restrict__ part) {
  __shared__ float red[256];
  const int row = blockIdx.y;
  const int ch  = blockIdx.x;                // 0..15
  const int CH  = 131072 / 16;               // 8192 floats
  const float4* W4 = (const float4*)(Wm + (size_t)row * 131072 + (size_t)ch * CH);
  const float4* X4 = (const float4*)(xv + (size_t)ch * CH);
  int t = threadIdx.x;
  float s = 0.f;
#pragma unroll
  for (int j = 0; j < 8; ++j) {
    int idx = t + j * 256;
    __builtin_prefetch((const void*)(W4 + idx + 512), 0, 1);
    float4 w = W4[idx], xx = X4[idx];
    s += w.x * xx.x + w.y * xx.y + w.z * xx.z + w.w * xx.w;
  }
  red[t] = s;
  __syncthreads();
  for (int o = 128; o > 0; o >>= 1) {
    if (t < o) red[t] += red[t + o];
    __syncthreads();
  }
  if (t == 0) part[row * 16 + ch] = red[0];
}

__global__ void k_fc1_red(const float* __restrict__ part, const float* __restrict__ b1,
                          float* __restrict__ fc1) {
  int i = blockIdx.x * 256 + threadIdx.x;
  if (i < 1024) {
    float s = b1[i];
#pragma unroll
    for (int c = 0; c < 16; ++c) s += part[i * 16 + c];
    fc1[i] = s;
  }
}

__global__ void k_fc2(const float* __restrict__ W2, const float* __restrict__ b2,
                      const float* __restrict__ h, float* __restrict__ out) {
  int t = threadIdx.x, w = t >> 5, lane = t & 31;
  float s = 0.f;
#pragma unroll
  for (int j = 0; j < 32; ++j) s += W2[w * 1024 + lane + j * 32] * h[lane + j * 32];
  for (int o = 16; o > 0; o >>= 1) s += __shfl_down(s, o, 32);
  if (lane == 0) out[w] = s + b2[w];
}

// ---------------------------------------------------------------------------
// Host orchestration
// ---------------------------------------------------------------------------
extern "C" void kernel_launch(void* const* d_in, const int* in_sizes, int n_in,
                              void* d_out, int out_size, void* d_ws, size_t ws_size,
                              hipStream_t stream) {
  (void)in_sizes; (void)n_in; (void)out_size; (void)ws_size;
  auto Wp = [&](int i) { return (const float*)d_in[i]; };

  char* base = (char*)d_ws;
  size_t off = 0;
  auto alloc = [&](size_t bytes) -> char* {
    size_t a = (off + 255) & ~(size_t)255;
    off = a + bytes;
    return base + a;
  };
  _Float16* grayA = (_Float16*)alloc((size_t)130 * 130 * 32 * 2);
  _Float16* grayB = (_Float16*)alloc((size_t)130 * 130 * 32 * 2);
  _Float16* bufA  = (_Float16*)alloc((size_t)130 * 130 * 64 * 2);
  _Float16* bufB  = (_Float16*)alloc((size_t)130 * 130 * 64 * 2);
  _Float16* fa_n  = (_Float16*)alloc((size_t)256 * 64 * 2);
  _Float16* fbp   = (_Float16*)alloc((size_t)48 * 48 * 64 * 2);
  _Float16* cvb   = (_Float16*)alloc((size_t)18 * 18 * 1120 * 2);
  _Float16* estA  = (_Float16*)alloc((size_t)18 * 18 * 512 * 2);
  _Float16* estB  = (_Float16*)alloc((size_t)18 * 18 * 512 * 2);
  float*    xf    = (float*)alloc((size_t)131072 * 4);
  float*    fc1   = (float*)alloc(1024 * 4);
  float*    part  = (float*)alloc(1024 * 16 * 4);
  _Float16* wfoldA = (_Float16*)alloc((size_t)64 * 288 * 2);
  _Float16* wfoldB = (_Float16*)alloc((size_t)64 * 288 * 2);
  _Float16* wExtA  = (_Float16*)alloc((size_t)7 * 64 * 576 * 2);
  _Float16* wExtB  = (_Float16*)alloc((size_t)7 * 64 * 576 * 2);
  _Float16* wcv1   = (_Float16*)alloc((size_t)512 * 10080 * 2);
  _Float16* wcv2   = (_Float16*)alloc((size_t)512 * 4608 * 2);
  _Float16* wcv3   = (_Float16*)alloc((size_t)512 * 4608 * 2);

  auto zero = [&](_Float16* p, int n) {
    k_zero_h<<<(n + 255) / 256, 256, 0, stream>>>(p, n);
  };
  auto conv = [&](const _Float16* in, const _Float16* wt, const float* bias,
                  _Float16* outh, float* outf, int H, int Wd, int Cin, int Cout, int relu) {
    if (outh) zero(outh, (H + 2) * (Wd + 2) * Cout);
    int tiles  = (Cout / 16) * (H * Wd / 16);
    k_conv3x3_wmma<<<(tiles + 7) / 8, 256, 0, stream>>>(in, wt, bias, outh, outf,
                                                        H, Wd, Cin, Cout, relu);
  };
  auto pool = [&](const _Float16* in, _Float16* out, int C, int Ho, int Wo) {
    zero(out, (Ho + 2) * (Wo + 2) * C);
    int tot = Ho * Wo * C;
    k_maxpool<<<(tot + 255) / 256, 256, 0, stream>>>(in, out, C, Ho, Wo);
  };

  // feature extractor: params base bi (2 for A, 18 for B); result padded 18x18x64 in bufA
  auto extract = [&](int bi, _Float16* gray, _Float16* wfold, _Float16* wl) -> _Float16* {
    k_fold<<<(64 * 288 + 255) / 256, 256, 0, stream>>>(Wp(bi + 0), wfold);
    for (int l = 0; l < 7; ++l)
      k_cvt<<<(64 * 576 + 255) / 256, 256, 0, stream>>>(Wp(bi + 2 + 2 * l),
                                                        wl + (size_t)l * 64 * 576, 64, 64, 64);
    conv(gray, wfold, Wp(bi + 1), bufA, nullptr, 128, 128, 32, 64, 1);           // 1a folded
    conv(bufA, wl + 0 * 36864, Wp(bi + 3), bufB, nullptr, 128, 128, 64, 64, 1);  // 1b
    pool(bufB, bufA, 64, 64, 64);
    conv(bufA, wl + 1 * 36864, Wp(bi + 5), bufB, nullptr, 64, 64, 64, 64, 1);    // 2a
    conv(bufB, wl + 2 * 36864, Wp(bi + 7), bufA, nullptr, 64, 64, 64, 64, 1);    // 2b
    pool(bufA, bufB, 64, 32, 32);
    conv(bufB, wl + 3 * 36864, Wp(bi + 9), bufA, nullptr, 32, 32, 64, 64, 1);    // 3a
    conv(bufA, wl + 4 * 36864, Wp(bi + 11), bufB, nullptr, 32, 32, 64, 64, 1);   // 3b
    pool(bufB, bufA, 64, 16, 16);
    conv(bufA, wl + 5 * 36864, Wp(bi + 13), bufB, nullptr, 16, 16, 64, 64, 1);   // 4a
    conv(bufB, wl + 6 * 36864, Wp(bi + 15), bufA, nullptr, 16, 16, 64, 64, 1);   // 4b
    return bufA;
  };

  // ---- pipeline ----
  zero(grayA, 130 * 130 * 32);
  zero(grayB, 130 * 130 * 32);
  k_gray<<<64, 256, 0, stream>>>((const float*)d_in[0], grayA);
  k_gray<<<64, 256, 0, stream>>>((const float*)d_in[1], grayB);
  zero(fbp, 48 * 48 * 64);

  _Float16* fA = extract(2, grayA, wfoldA, wExtA);
  k_l2norm<<<1, 256, 0, stream>>>(fA, fa_n, 0);
  _Float16* fB = extract(18, grayB, wfoldB, wExtB);
  k_l2norm<<<1, 256, 0, stream>>>(fB, fbp, 1);

  zero(cvb, 18 * 18 * 1120);
  k_costvol<<<1089, 256, 0, stream>>>(fa_n, fbp, cvb);

  k_cvt<<<((512 * 10080) + 255) / 256, 256, 0, stream>>>(Wp(34), wcv1, 512, 1089, 1120);
  conv(cvb, wcv1, Wp(35), estA, nullptr, 16, 16, 1120, 512, 1);
  k_cvt<<<((512 * 4608) + 255) / 256, 256, 0, stream>>>(Wp(36), wcv2, 512, 512, 512);
  conv(estA, wcv2, Wp(37), estB, nullptr, 16, 16, 512, 512, 1);
  k_cvt<<<((512 * 4608) + 255) / 256, 256, 0, stream>>>(Wp(38), wcv3, 512, 512, 512);
  conv(estB, wcv3, Wp(39), nullptr, xf, 16, 16, 512, 512, 1);   // f32 NCHW flatten order

  k_fc1<<<dim3(16, 1024), 256, 0, stream>>>(Wp(40), xf, part);
  k_fc1_red<<<4, 256, 0, stream>>>(part, Wp(41), fc1);
  k_fc2<<<1, 256, 0, stream>>>(Wp(42), Wp(43), fc1, (float*)d_out);
}
